// MLPDecoder_79224966742230
// MI455X (gfx1250) — matrix operand order repaired and checked by reading, not compiled, not run on hardware
//
#include <hip/hip_runtime.h>
#include <hip/hip_bf16.h>

typedef __attribute__((ext_vector_type(16))) _Float16 v16h;
typedef __attribute__((ext_vector_type(8)))  _Float16 v8h;
typedef __attribute__((ext_vector_type(4)))  _Float16 v4h;
typedef __attribute__((ext_vector_type(8)))  float    v8f;

#define D_FEAT   128
#define N_EDGES  500000
#define HIDDEN   256
#define KDIM     256                // 2*D_FEAT
#define KROW     264                // padded LDS row (halves): 528B stride -> bank-conflict free
#define NTILES   (N_EDGES / 16)     // 31250
#define NW       8                  // waves per block
#define BLOCK    (NW * 32)          // 256 threads

#define WGT_BYTES (HIDDEN * KROW * 2)        // 135168
#define B1_OFF    (WGT_BYTES)                // 135168
#define W2_OFF    (B1_OFF + 1024)            // 136192
#define AT_OFF    (W2_OFF + 1024)            // 137216
#define AT_BYTES  (16 * KROW * 2)            // 8448
#define SMEM_BYTES (AT_OFF + NW * AT_BYTES)  // 204800 (<= 320KB WGP LDS)

__global__ __launch_bounds__(BLOCK) void edge_mlp_wmma_kernel(
    const float* __restrict__ z,
    const long long* __restrict__ ei,
    const float* __restrict__ fc1w,
    const float* __restrict__ fc1b,
    const float* __restrict__ fc2w,
    const float* __restrict__ fc2b,
    float* __restrict__ out)
{
  extern __shared__ char smem[];
  _Float16* wgt = (_Float16*)(smem);
  float*    b1s = (float*)(smem + B1_OFF);
  float*    w2s = (float*)(smem + W2_OFF);

  const int tid  = threadIdx.x;
  const int wid  = tid >> 5;
  const int lane = tid & 31;
  _Float16* aT = (_Float16*)(smem + AT_OFF + wid * AT_BYTES);

  // ---- stage fc1_w (f32->f16, padded rows), fc1_b, fc2_w into LDS ----
  {
    const float4* src4 = (const float4*)fc1w;
    for (int i = tid; i < HIDDEN * (KDIM / 4); i += BLOCK) {
      int h = i >> 6, k4 = i & 63;
      float4 f = src4[i];
      v4h hv;
      hv[0] = (_Float16)f.x; hv[1] = (_Float16)f.y;
      hv[2] = (_Float16)f.z; hv[3] = (_Float16)f.w;
      *(v4h*)(wgt + h * KROW + k4 * 4) = hv;
    }
  }
  for (int i = tid; i < HIDDEN; i += BLOCK) {
    b1s[i] = fc1b[i];
    w2s[i] = fc2w[i];
  }
  __syncthreads();

  const float bias2 = fc2b[0];
  const int hl   = lane & 15;   // A-row / B-col / C-col position of this lane
  const int half = lane >> 4;   // K-half selector (WMMA 16-bit layouts)

  for (int tile = blockIdx.x * NW + wid; tile < NTILES; tile += gridDim.x * NW) {
    // ---- gather 16 edges' concat(z[row], z[col]) into per-wave LDS tile (f32->f16) ----
    {
      const int e    = lane >> 1;                 // edge within tile (2 lanes/edge)
      const int hsel = lane & 1;                  // 0: z[row] half, 1: z[col] half
      const long long base = (long long)(tile * 16 + e) * 3;
      const long long idx  = ei[base + 1 + hsel]; // columns 1,2 of edge_index
      const float4* src = (const float4*)(z + idx * D_FEAT);
      v8h* dst = (v8h*)(aT + e * KROW + hsel * D_FEAT);
#pragma unroll 4
      for (int i = 0; i < 16; ++i) {
        float4 f0 = src[2 * i + 0];
        float4 f1 = src[2 * i + 1];
        v8h h;
        h[0] = (_Float16)f0.x; h[1] = (_Float16)f0.y;
        h[2] = (_Float16)f0.z; h[3] = (_Float16)f0.w;
        h[4] = (_Float16)f1.x; h[5] = (_Float16)f1.y;
        h[6] = (_Float16)f1.z; h[7] = (_Float16)f1.w;
        dst[i] = h;
      }
    }
    // per-wave LDS RAW fence (DS ops are in-order per wave; be explicit)
    asm volatile("s_wait_dscnt 0" ::: "memory");

    // ---- FC1 tile: X[16x256] = A[16x256] @ W1^T[256x256], f16 WMMA, f32 accum ----
    v8f acc[16];
#pragma unroll
    for (int nt = 0; nt < 16; ++nt) acc[nt] = (v8f){0.f,0.f,0.f,0.f,0.f,0.f,0.f,0.f};

    // keep this a real loop: acc[16] pins 128 VGPRs for the whole K sweep, so
    // limit concurrently-live A/B fragments to avoid scratch spills
#pragma unroll 1
    for (int ks = 0; ks < 8; ++ks) {
      // A fragment (16x32, 16-bit): lane<16 holds K {0..7,16..23}, lane>=16 {8..15,24..31}
      const int ka = ks * 32 + half * 8;
      v8h a0 = *(const v8h*)(aT + hl * KROW + ka);
      v8h a1 = *(const v8h*)(aT + hl * KROW + ka + 16);
      v16h a = __builtin_shufflevector(a0, a1, 0,1,2,3,4,5,6,7,8,9,10,11,12,13,14,15);
      // B fragment (32x16, 16-bit): lane = column N, same K order as A
      const int kb = ks * 32 + half * 8;
#pragma unroll
      for (int nt = 0; nt < 16; ++nt) {
        const _Float16* brow = wgt + (nt * 16 + hl) * KROW + kb;
        v8h bq0 = *(const v8h*)(brow);
        v8h bq1 = *(const v8h*)(brow + 16);
        v16h b = __builtin_shufflevector(bq0, bq1, 0,1,2,3,4,5,6,7,8,9,10,11,12,13,14,15);
        acc[nt] = __builtin_amdgcn_wmma_f32_16x16x32_f16(
            false, a, false, b, (short)0, acc[nt], false, false);
      }
    }

    // ---- epilogue: relu(x + b1) . fc2_w, + fc2_b, sigmoid ----
    // C layout: lane L, vgpr j -> edge M = j + 8*(L>=16), hidden h = nt*16 + (L%16)
    float partial[8] = {0.f,0.f,0.f,0.f,0.f,0.f,0.f,0.f};
#pragma unroll
    for (int nt = 0; nt < 16; ++nt) {
      const int h = nt * 16 + hl;
      const float b1 = b1s[h];
      const float w2 = w2s[h];
#pragma unroll
      for (int j = 0; j < 8; ++j) {
        float x = acc[nt][j] + b1;
        x = x > 0.f ? x : 0.f;
        partial[j] += x * w2;
      }
    }
    // reduce across the 16 lanes of each half (xor 1,2,4,8 stays within the half)
#pragma unroll
    for (int off = 1; off < 16; off <<= 1) {
#pragma unroll
      for (int j = 0; j < 8; ++j)
        partial[j] += __shfl_xor(partial[j], off, 32);
    }
    if (hl == 0) {
      const int eb = tile * 16 + half * 8;
#pragma unroll
      for (int j = 0; j < 8; ++j) {
        float s = partial[j] + bias2;
        out[eb + j] = 1.f / (1.f + __expf(-s));
      }
    }
  }
}

extern "C" void kernel_launch(void* const* d_in, const int* in_sizes, int n_in,
                              void* d_out, int out_size, void* d_ws, size_t ws_size,
                              hipStream_t stream) {
  const float*     z    = (const float*)d_in[0];
  const long long* ei   = (const long long*)d_in[1];
  const float*     fc1w = (const float*)d_in[2];
  const float*     fc1b = (const float*)d_in[3];
  const float*     fc2w = (const float*)d_in[4];
  const float*     fc2b = (const float*)d_in[5];
  float*           out  = (float*)d_out;

  (void)in_sizes; (void)n_in; (void)out_size; (void)d_ws; (void)ws_size;

  // allow > 64KB dynamic LDS (host-side attribute set; idempotent, capture-safe)
  hipFuncSetAttribute((const void*)edge_mlp_wmma_kernel,
                      hipFuncAttributeMaxDynamicSharedMemorySize, SMEM_BYTES);

  dim3 grid(1024), block(BLOCK);
  edge_mlp_wmma_kernel<<<grid, block, SMEM_BYTES, stream>>>(
      z, ei, fc1w, fc1b, fc2w, fc2b, out);
}
